// Sot_41008347743021
// MI455X (gfx1250) — matrix operand order repaired and checked by reading, not compiled, not run on hardware
//
#include <hip/hip_runtime.h>

#define N 4096
#define DIM 640

typedef __attribute__((ext_vector_type(16))) _Float16 v16h;
typedef __attribute__((ext_vector_type(8)))  _Float16 v8h;
typedef __attribute__((ext_vector_type(8)))  float    v8f;

union Frag {
  v16h v;
  v8h  h[2];
};

// ---------------------------------------------------------------------------
// Kernel 1: row-normalize x (f32) -> V (f16). One block per row.
// ---------------------------------------------------------------------------
__global__ void __launch_bounds__(256)
rownorm_kernel(const float* __restrict__ x, _Float16* __restrict__ vh) {
  __shared__ float red[256];
  const int row = blockIdx.x;
  const float* xr = x + (size_t)row * DIM;
  float s = 0.0f;
  for (int j = threadIdx.x; j < DIM; j += 256) {
    float t = xr[j];
    s += t * t;
  }
  red[threadIdx.x] = s;
  __syncthreads();
  for (int off = 128; off > 0; off >>= 1) {
    if (threadIdx.x < off) red[threadIdx.x] += red[threadIdx.x + off];
    __syncthreads();
  }
  const float rinv = rsqrtf(red[0]);
  _Float16* vr = vh + (size_t)row * DIM;
  for (int j = threadIdx.x; j < DIM; j += 256) {
    vr[j] = (_Float16)(xr[j] * rinv);
  }
}

// ---------------------------------------------------------------------------
// Kernel 2: K = exp(0.2*(V V^T - 1)), diag = 0.  WMMA f32_16x16x32_f16.
// 256 threads = 8 waves arranged 2(M) x 4(N); wave tile 64x32 (4x2 WMMA
// tiles -> 8 WMMA per k-step against 12 b128 loads); block tile 128x128.
// ---------------------------------------------------------------------------
__global__ void __launch_bounds__(256)
gemm_exp_kernel(const _Float16* __restrict__ vh, float* __restrict__ K) {
  const int wave  = threadIdx.x >> 5;
  const int lane  = threadIdx.x & 31;
  const int lhalf = lane >> 4;     // 0 or 1
  const int l16   = lane & 15;

  const int m0 = blockIdx.x * 128 + (wave & 1) * 64;   // 4 tiles of 16 in M
  const int n0 = blockIdx.y * 128 + (wave >> 1) * 32;  // 2 tiles of 16 in N

  v8f acc[4][2];
  #pragma unroll
  for (int i = 0; i < 4; ++i)
    #pragma unroll
    for (int j = 0; j < 2; ++j) acc[i][j] = (v8f){};

  // A fragment: lane holds row (m + l16), K interleaved in blocks of 8 per half-wave
  const _Float16* arow[4];
  #pragma unroll
  for (int i = 0; i < 4; ++i) arow[i] = vh + (size_t)(m0 + 16 * i + l16) * DIM;
  // B fragment: column n of V^T = row n of V (contiguous); K interleaved in blocks of 8 per half-wave
  const _Float16* brow[2];
  #pragma unroll
  for (int j = 0; j < 2; ++j) brow[j] = vh + (size_t)(n0 + 16 * j + l16) * DIM;

  const int aoff = lhalf * 8;
  const int boff = lhalf * 8;

  for (int kb = 0; kb < DIM; kb += 32) {
    Frag a[4], b[2];
    #pragma unroll
    for (int j = 0; j < 2; ++j) {
      b[j].h[0] = *(const v8h*)(brow[j] + kb + boff);
      b[j].h[1] = *(const v8h*)(brow[j] + kb + boff + 16);
    }
    #pragma unroll
    for (int i = 0; i < 4; ++i) {
      a[i].h[0] = *(const v8h*)(arow[i] + kb + aoff);
      a[i].h[1] = *(const v8h*)(arow[i] + kb + aoff + 16);
    }
    #pragma unroll
    for (int i = 0; i < 4; ++i)
      #pragma unroll
      for (int j = 0; j < 2; ++j)
        acc[i][j] = __builtin_amdgcn_wmma_f32_16x16x32_f16(
            false, a[i].v, false, b[j].v, (short)0, acc[i][j], false, false);
  }

  // Epilogue: K_ij = exp(0.2*(S_ij - 1)); diagonal forced to 0.
  // C layout: VGPR t -> row = mt + t + 8*lhalf ; col = nt + l16
  #pragma unroll
  for (int ti = 0; ti < 4; ++ti) {
    #pragma unroll
    for (int tj = 0; tj < 2; ++tj) {
      const int mt  = m0 + ti * 16;
      const int col = n0 + tj * 16 + l16;
      #pragma unroll
      for (int t = 0; t < 8; ++t) {
        const int row = mt + t + 8 * lhalf;
        float s = acc[ti][tj][t];
        float k = __expf(0.2f * (s - 1.0f));
        if (row == col) k = 0.0f;
        K[(size_t)row * N + col] = k;
      }
    }
  }
}

// ---------------------------------------------------------------------------
// Kernel 3: u = 1/N
// ---------------------------------------------------------------------------
__global__ void init_kernel(float* __restrict__ u) {
  int i = blockIdx.x * 256 + threadIdx.x;
  if (i < N) u[i] = 1.0f / (float)N;
}

// ---------------------------------------------------------------------------
// Kernel 4: out[r] = 1 / (K[r,:] . xin).  K symmetric -> also serves K^T.
// One block per row; float4 streaming (K lives in L2, 64MB < 192MB).
// ---------------------------------------------------------------------------
__global__ void __launch_bounds__(256)
matvec_recip_kernel(const float* __restrict__ K, const float* __restrict__ xin,
                    float* __restrict__ xout) {
  __shared__ float red[256];
  const int row = blockIdx.x;
  const float4* kr = (const float4*)(K + (size_t)row * N);
  const float4* xv = (const float4*)xin;
  float s = 0.0f;
  for (int j = threadIdx.x; j < N / 4; j += 256) {
    float4 kv = kr[j];
    float4 x4 = xv[j];
    s += kv.x * x4.x + kv.y * x4.y + kv.z * x4.z + kv.w * x4.w;
  }
  red[threadIdx.x] = s;
  __syncthreads();
  for (int off = 128; off > 0; off >>= 1) {
    if (threadIdx.x < off) red[threadIdx.x] += red[threadIdx.x + off];
    __syncthreads();
  }
  if (threadIdx.x == 0) xout[row] = 1.0f / red[0];
}

// ---------------------------------------------------------------------------
// Kernel 5: in-place W = u_i * K_ij * v_j, diag = 1.  float4 per thread.
// ---------------------------------------------------------------------------
__global__ void __launch_bounds__(256)
finalize_kernel(float* __restrict__ K, const float* __restrict__ u,
                const float* __restrict__ v) {
  const size_t t    = (size_t)blockIdx.x * 256 + threadIdx.x;
  const size_t base = t * 4;
  const int i = (int)(base >> 12);   // / 4096
  const int j = (int)(base & 4095);
  const float ui = u[i];
  float4 kv = *(float4*)(K + base);
  float4 w;
  w.x = ui * kv.x * v[j + 0];
  w.y = ui * kv.y * v[j + 1];
  w.z = ui * kv.z * v[j + 2];
  w.w = ui * kv.w * v[j + 3];
  const int d = i - j;               // diagonal lands at element d if 0<=d<4
  if (d >= 0 && d < 4) ((float*)&w)[d] = 1.0f;
  *(float4*)(K + base) = w;
}

// ---------------------------------------------------------------------------
extern "C" void kernel_launch(void* const* d_in, const int* in_sizes, int n_in,
                              void* d_out, int out_size, void* d_ws, size_t ws_size,
                              hipStream_t stream) {
  const float* x = (const float*)d_in[0];
  float* Kmat = (float*)d_out;             // K lives in d_out, rescaled in place

  char* ws = (char*)d_ws;
  _Float16* vh = (_Float16*)ws;                        // N*DIM f16 = 5.25 MB
  float* u = (float*)(ws + (size_t)N * DIM * sizeof(_Float16));
  float* v = u + N;

  rownorm_kernel<<<N, 256, 0, stream>>>(x, vh);

  dim3 grid(N / 128, N / 128);
  gemm_exp_kernel<<<grid, 256, 0, stream>>>(vh, Kmat);

  init_kernel<<<N / 256, 256, 0, stream>>>(u);

  for (int it = 0; it < 10; ++it) {
    matvec_recip_kernel<<<N, 256, 0, stream>>>(Kmat, u, v);  // v = 1/(K^T u)
    matvec_recip_kernel<<<N, 256, 0, stream>>>(Kmat, v, u);  // u = 1/(K v)
  }

  finalize_kernel<<<(N / 4) * (N / 256), 256, 0, stream>>>(Kmat, u, v);
}
